// Gen_48790828483054
// MI455X (gfx1250) — matrix operand order repaired and checked by reading, not compiled, not run on hardware
//
#include <hip/hip_runtime.h>
#include <hip/hip_bf16.h>

// Problem constants (from the reference)
#define TT 6
#define VV 31
#define LL 3

typedef _Float16 v16h __attribute__((ext_vector_type(16)));
typedef _Float16 v8h  __attribute__((ext_vector_type(8)));
typedef _Float16 h2   __attribute__((ext_vector_type(2)));
typedef float    v8f  __attribute__((ext_vector_type(8)));
typedef float    f2   __attribute__((ext_vector_type(2)));

// LDS tile strides (padded vs 32 to spread banks; keep 16B/8B alignment)
#define HSTR 48   // f16 row stride (96 B)
#define FSTR 34   // f32 row stride (136 B, keeps float2 stores 8B-aligned)

// Output-column permutation: N-tile0 = even features, N-tile1 = odd features.
// Slot s in tile0 -> feature 2s ; slot s in tile1 -> feature 2s+1.
__host__ __device__ __forceinline__ int perm_col(int n) {
  return (n & 1) * 16 + (n >> 1);
}

__device__ __forceinline__ float ftanh(float x) {
#if __has_builtin(__builtin_amdgcn_tanhf)
  return __builtin_amdgcn_tanhf(x);       // v_tanh_f32 (gfx1250 TRANS op)
#else
  float r;
  asm volatile("v_tanh_f32 %0, %1\n\tv_nop" : "=v"(r) : "v"(x));
  return r;
#endif
}

__device__ __forceinline__ v8f wmma_f16(v16h a, v16h b, v8f c) {
  // (neg_a, A, neg_b, B, c_mod, C, reuse_a, reuse_b)
  return __builtin_amdgcn_wmma_f32_16x16x32_f16(false, a, false, b, (short)0, c,
                                                false, false);
}

__device__ __forceinline__ void wait_ds0() {
  asm volatile("s_wait_dscnt 0" ::: "memory");
}

// ---------------------------------------------------------------------------
// Prep: pack weights as f16 in WMMA B-fragment layout.
//  * even/odd column permutation -> D pairs (c0[r],c1[r]) are feature-adjacent
//  * bias folded into the K=31 pad row (activations carry 1.0 at K=31), so
//    WMMA accumulators start from the inline-0 SRC2 operand (no bias movs)
//   wpack[l][k][perm(n)]: k in 0..30  = Wih[l][n][k]
//                         k == 31     = b_ih[l][n] + b_hh[l][n]   (bias row)
//                         k in 32..62 = Whh[l][n][k-32] ; k == 63 = 0
//   lpack[k][perm(n)]   : k in 0..30 = Wlin[n][k] ; k == 31 = b_lin[n]
// ---------------------------------------------------------------------------
__global__ void rnn_prep(const float* __restrict__ Wih, const float* __restrict__ Whh,
                         const float* __restrict__ bih, const float* __restrict__ bhh,
                         const float* __restrict__ Wlin, const float* __restrict__ blin,
                         _Float16* __restrict__ wpack, _Float16* __restrict__ lpack) {
  const int tid = threadIdx.x;
  for (int idx = tid; idx < LL * 64 * 32; idx += blockDim.x) {
    int l = idx / 2048, r = idx % 2048, k = r >> 5, n = r & 31;
    float v = 0.f;
    if (n < VV) {
      if (k < VV)                    v = Wih[(l * VV + n) * VV + k];
      else if (k == VV)              v = bih[l * VV + n] + bhh[l * VV + n];
      else if (k >= 32 && k < 32+VV) v = Whh[(l * VV + n) * VV + (k - 32)];
    }
    wpack[(l * 64 + (k & 32) + ((k >> 3) & 1) * 16 + (perm_col(n) & 15)) * 32 + (perm_col(n) & 16) + (k & 7) + ((k >> 4) & 1) * 8] = (_Float16)v;
  }
  for (int idx = tid; idx < 32 * 32; idx += blockDim.x) {
    int k = idx >> 5, n = idx & 31;
    float v = 0.f;
    if (n < VV) {
      if (k < VV)       v = Wlin[n * VV + k];
      else if (k == VV) v = blin[n];
    }
    lpack[(((k >> 3) & 1) * 16 + (perm_col(n) & 15)) * 32 + (perm_col(n) & 16) + (k & 7) + ((k >> 4) & 1) * 8] = (_Float16)v;
  }
}

// ---------------------------------------------------------------------------
// Main: one wave per 16 batch rows. x[t] held as f16 WMMA A-fragments in VGPRs
// with a constant 1.0 in the K=31 pad slot (multiplies the bias row).
// Per (layer,t): D = tanh( [x|1|h] * [Wih;b;Whh;0]^T ) via 4 WMMAs from C=0.
// D->A relayout through a private LDS tile (4x ds_store_2addr_b32 +
// 2x ds_load_b128 per step, in-wave s_wait_dscnt only).
// ---------------------------------------------------------------------------
__global__ void __launch_bounds__(256)
rnn_wmma(const float* __restrict__ inp, const float* __restrict__ h0,
         const _Float16* __restrict__ wpack, const _Float16* __restrict__ lpack,
         float* __restrict__ out, int Bn) {
  __shared__ __align__(32) _Float16 lds_h[8][16 * HSTR];
  __shared__ __align__(16) float    lds_f[8][16 * FSTR];

  const int lane = threadIdx.x & 31;
  const int wave = threadIdx.x >> 5;
  const int tile = blockIdx.x * 8 + wave;
  const int row0 = tile * 16;
  if (row0 >= Bn) return;                 // wave-uniform exit, EXEC stays full

  const int m = lane & 15;                // A-fragment: M = lane % 16
  const int g = lane >> 4;                // lane group selects K sub-blocks
  int gb = row0 + m; if (gb >= Bn) gb = Bn - 1;   // clamp loads (full EXEC)

  // --- load x[t] A-fragments (f32 -> f16), ISA 16-bit A layout:
  //     elems 0..7 -> K = g*8 + j ; elems 8..15 -> K = 16 + g*8 + (j-8)
  //     K==31 (g=1, elem 15) carries the constant 1.0 for the bias row.
  v16h x[TT];
  #pragma unroll
  for (int t = 0; t < TT; ++t) {
    const float* p = inp + ((size_t)gb * TT + t) * VV;
    #pragma unroll
    for (int j = 0; j < 16; ++j) {
      int k = (j < 8) ? (g * 8 + j) : (16 + g * 8 + (j - 8));
      float v = (k < VV) ? p[k] : ((k == VV) ? 1.0f : 0.0f);
      x[t][j] = (_Float16)v;
    }
  }

  _Float16* tb = lds_h[wave];
  const int colA  = lane & 15;            // C/D layout: N(slot) = lane % 16
  const int rbase = g * 8;                // C/D layout: M = r + 8*(lane/16)

  for (int l = 0; l < LL; ++l) {
    // B-fragments: lane%16 = column slot, lane/16 picks K sub-blocks, 16 contiguous K values (32B aligned)
    const _Float16* wl = wpack + l * 64 * 32;
    v16h wx0 = *reinterpret_cast<const v16h*>(wl + lane * 32);
    v16h wx1 = *reinterpret_cast<const v16h*>(wl + lane * 32 + 16);
    v16h wh0 = *reinterpret_cast<const v16h*>(wl + (32 + lane) * 32);
    v16h wh1 = *reinterpret_cast<const v16h*>(wl + (32 + lane) * 32 + 16);

    // h0 A-fragment for this layer (pad slot multiplies zero row 63 -> 0 ok)
    v16h h;
    const float* hp = h0 + ((size_t)l * Bn + gb) * VV;
    #pragma unroll
    for (int j = 0; j < 16; ++j) {
      int k = (j < 8) ? (g * 8 + j) : (16 + g * 8 + (j - 8));
      h[j] = (k < VV) ? (_Float16)hp[k] : (_Float16)0.f;
    }

    #pragma unroll
    for (int t = 0; t < TT; ++t) {
      v8f c0 = {}, c1 = {};               // inline-0 SRC2: bias comes via K=31
      c0 = wmma_f16(x[t], wx0, c0);
      c0 = wmma_f16(h,    wh0, c0);
      c1 = wmma_f16(x[t], wx1, c1);
      c1 = wmma_f16(h,    wh1, c1);

      // tanh + packed store: (c0[r], c1[r]) = features (2*colA, 2*colA+1)
      #pragma unroll
      for (int r = 0; r < 8; ++r) {
        h2 pk;
        pk[0] = (_Float16)ftanh(c0[r]);
        pk[1] = (_Float16)ftanh(c1[r]);
        *reinterpret_cast<h2*>(tb + (rbase + r) * HSTR + 2 * colA) = pk;
      }
      wait_ds0();  // in-wave lockstep: LDS writes visible, no block barrier
      // gather back as A-fragment: two 16B contiguous LDS loads
      v8h lo = *reinterpret_cast<const v8h*>(tb + m * HSTR + g * 8);
      v8h hi = *reinterpret_cast<const v8h*>(tb + m * HSTR + 16 + g * 8);
      v16h nf;
      #pragma unroll
      for (int j = 0; j < 8; ++j) { nf[j] = lo[j]; nf[8 + j] = hi[j]; }
      nf[15] = g ? (_Float16)1.0f : nf[15];   // restore 1.0 in the K=31 slot
      x[t] = nf;
      h    = nf;
    }
  }

  // --- final linear + tanh on x[T-1] (bias via K=31 row of lpack)
  v16h wl0 = *reinterpret_cast<const v16h*>(lpack + lane * 32);
  v16h wl1 = *reinterpret_cast<const v16h*>(lpack + lane * 32 + 16);
  v8f c0 = {}, c1 = {};
  c0 = wmma_f16(x[TT - 1], wl0, c0);
  c1 = wmma_f16(x[TT - 1], wl1, c1);

  float* fb = lds_f[wave];
  #pragma unroll
  for (int r = 0; r < 8; ++r) {
    f2 pk;
    pk[0] = ftanh(c0[r]);
    pk[1] = ftanh(c1[r]);
    *reinterpret_cast<f2*>(fb + (rbase + r) * FSTR + 2 * colA) = pk;  // b64 store
  }
  wait_ds0();

  // --- softmax: one batch row per lane (lanes 0..15), natural feature order
  if (lane < 16) {
    const int orow = row0 + lane;
    if (orow < Bn) {
      const float* rp = fb + lane * FSTR;
      float mx = rp[0];
      #pragma unroll
      for (int n = 1; n < VV; ++n) mx = fmaxf(mx, rp[n]);
      float e[VV]; float s = 0.f;
      #pragma unroll
      for (int n = 0; n < VV; ++n) { e[n] = __expf(rp[n] - mx); s += e[n]; }
      const float inv = 1.0f / s;
      float* op = out + (size_t)orow * VV;
      #pragma unroll
      for (int n = 0; n < VV; ++n) op[n] = e[n] * inv;
    }
  }
}

extern "C" void kernel_launch(void* const* d_in, const int* in_sizes, int n_in,
                              void* d_out, int out_size, void* d_ws, size_t ws_size,
                              hipStream_t stream) {
  const float* inp  = (const float*)d_in[0];
  const float* h0   = (const float*)d_in[1];
  const float* Wih  = (const float*)d_in[2];
  const float* Whh  = (const float*)d_in[3];
  const float* bih  = (const float*)d_in[4];
  const float* bhh  = (const float*)d_in[5];
  const float* Wlin = (const float*)d_in[6];
  const float* blin = (const float*)d_in[7];
  float* out = (float*)d_out;

  const int Bn = in_sizes[0] / (TT * VV);

  // Workspace layout (all offsets 32B-aligned for v16h loads)
  char* ws = (char*)d_ws;
  _Float16* wpack = (_Float16*)(ws);            // 3*64*32 f16 = 12288 B
  _Float16* lpack = (_Float16*)(ws + 12288);    // 32*32 f16   =  2048 B

  rnn_prep<<<1, 256, 0, stream>>>(Wih, Whh, bih, bhh, Wlin, blin, wpack, lpack);

  const int tiles  = (Bn + 15) / 16;
  const int blocks = (tiles + 7) / 8;           // 8 waves/block, 16 rows/wave
  rnn_wmma<<<blocks, 256, 0, stream>>>(inp, h0, wpack, lpack, out, Bn);
}
